// FourierConvolution2D_28707561407320
// MI455X (gfx1250) — matrix operand order repaired and checked by reading, not compiled, not run on hardware
//
#include <hip/hip_runtime.h>

// FourierConvolution2D on MI455X (gfx1250): direct 3x3 conv as implicit GEMM
// using V_WMMA_F32_16X16X32_BF16 with hi/lo bf16 split of fp32 data
// (3 WMMAs per product tile: hi*hi + hi*lo + lo*hi; lo*lo ~2^-18 dropped).

typedef __attribute__((ext_vector_type(8)))  float  v8f;
typedef __attribute__((ext_vector_type(8)))  __bf16 v8bf;
typedef __attribute__((ext_vector_type(16))) __bf16 v16bf;

#define BATCH 8
#define HH    64
#define WW    64
#define CIN   64
#define FF    64

// LDS layout (bf16 element strides).
// CSTR = 72 -> 144B row stride: 16B-aligned b128 loads, bank step 36 dwords
// (36*L mod 64 covers all 16 4-bank groups exactly once -> conflict-free).
#define CSTR   72
#define XROWS  4
#define XJ     68                    // jj = j+4, j in [-4, 63]
#define WELEMS (9 * FF * CSTR)       // per split: 41472 elems
#define XELEMS (XROWS * XJ * CSTR)   // per split: 19584 elems
#define LDS_ELEMS (2 * (WELEMS + XELEMS))

#define SHUF16(a, b) __builtin_shufflevector((a), (b), 0, 1, 2, 3, 4, 5, 6, 7, 8, 9, 10, 11, 12, 13, 14, 15)

__global__ __launch_bounds__(256) void fconv3x3_wmma_bf16x3(
    const float* __restrict__ x,     // (B,H,W,C) fp32
    const float* __restrict__ kern,  // (F,C,3,3) fp32
    const float* __restrict__ bias,  // (F,1,1)   fp32
    float* __restrict__ out)         // (B,H,W,F) fp32
{
    extern __shared__ __align__(16) __bf16 smem[];
    __bf16* Whi = smem;                    // [9][F][CSTR]
    __bf16* Wlo = smem + WELEMS;
    __bf16* Xhi = smem + 2 * WELEMS;       // [4][XJ][CSTR]
    __bf16* Xlo = Xhi + XELEMS;

    const int tid = threadIdx.x;
    const int blk = blockIdx.x;            // 0..255
    const int b   = blk >> 5;              // / (H/2)
    const int i0  = (blk & 31) << 1;       // 2 output rows per WG

    // ---- Stage weights: kern[o][c][mh][nw] -> W{hi,lo}[t = mh*3+nw][o][c]
    for (int idx = tid; idx < FF * CIN * 9; idx += 256) {
        const int o   = idx / (CIN * 9);
        const int rem = idx - o * (CIN * 9);
        const int c   = rem / 9;
        const int t   = rem - c * 9;
        const float w = kern[idx];
        const __bf16 h = (__bf16)w;
        const __bf16 l = (__bf16)(w - (float)h);
        const int off = (t * FF + o) * CSTR + c;
        Whi[off] = h;
        Wlo[off] = l;
    }

    // ---- Zero left halo (jj = 0..3)
    for (int idx = tid; idx < XROWS * 4 * CIN; idx += 256) {
        const int r   = idx / (4 * CIN);
        const int rem = idx - r * (4 * CIN);
        const int jj  = rem / CIN;
        const int c   = rem - jj * CIN;
        const int off = (r * XJ + jj) * CSTR + c;
        Xhi[off] = (__bf16)0.0f;
        Xlo[off] = (__bf16)0.0f;
    }

    // ---- Stage x rows i0-3 .. i0 (rows < 0 are zero)
    for (int idx = tid; idx < XROWS * WW * CIN; idx += 256) {
        const int r   = idx / (WW * CIN);
        const int rem = idx - r * (WW * CIN);
        const int j   = rem / CIN;
        const int c   = rem - j * CIN;
        const int xr  = i0 - 3 + r;
        float v = 0.0f;
        if (xr >= 0) v = x[(((b * HH) + xr) * WW + j) * CIN + c];
        const __bf16 h = (__bf16)v;
        const __bf16 l = (__bf16)(v - (float)h);
        const int off = (r * XJ + (j + 4)) * CSTR + c;
        Xhi[off] = h;
        Xlo[off] = l;
    }

    __syncthreads();

    // ---- Per-wave compute: 16 j positions x 64 output channels
    const int wave  = tid >> 5;
    const int lane  = tid & 31;
    const int laneM = lane & 15;           // tile row (A/M) or column (B/C/D, N)
    const int laneH = lane >> 4;           // lane half
    const int rbase = wave >> 2;           // 0 or 1 -> output row ir
    const int ir    = i0 + rbase;
    const int jbase = (wave & 3) << 4;

    v8f acc[4] = {};

#pragma unroll
    for (int t = 0; t < 9; ++t) {
        const int m = t / 3 + 1;
        const int n = t % 3 + 1;
        const int r = rbase + 3 - m;                 // LDS x-row, always 0..3
        const int jp = jbase + laneM + 4 - n;        // halo-shifted column
        const __bf16* arh = Xhi + (r * XJ + jp) * CSTR;
        const __bf16* arl = Xlo + (r * XJ + jp) * CSTR;
#pragma unroll
        for (int kh = 0; kh < 2; ++kh) {
            // A-matrix bf16 16x32 layout: lanes 0-15 hold K 0-7 & 16-23,
            // lanes 16-31 hold K 8-15 & 24-31 (absolute c = kh*32 + K).
            const int kb = kh * 32 + laneH * 8;
            const v8bf a0h = *(const v8bf*)(arh + kb);
            const v8bf a1h = *(const v8bf*)(arh + kb + 16);
            const v8bf a0l = *(const v8bf*)(arl + kb);
            const v8bf a1l = *(const v8bf*)(arl + kb + 16);
            const v16bf ah = SHUF16(a0h, a1h);
            const v16bf al = SHUF16(a0l, a1l);
#pragma unroll
            for (int ot = 0; ot < 4; ++ot) {
                // B-matrix 32x16: lane holds column o, K packed 2/dword;
                // lanes 0-15 -> K 0-7 & 16-23, lanes 16-31 -> K 8-15 & 24-31 of this slice.
                const int o  = (ot << 4) + laneM;
                const int cb = kh * 32 + laneH * 8;
                const __bf16* brh = Whi + (t * FF + o) * CSTR + cb;
                const __bf16* brl = Wlo + (t * FF + o) * CSTR + cb;
                const v8bf b0h = *(const v8bf*)(brh);
                const v8bf b1h = *(const v8bf*)(brh + 16);
                const v8bf b0l = *(const v8bf*)(brl);
                const v8bf b1l = *(const v8bf*)(brl + 16);
                const v16bf bh = SHUF16(b0h, b1h);
                const v16bf bl = SHUF16(b0l, b1l);
                acc[ot] = __builtin_amdgcn_wmma_f32_16x16x32_bf16(
                    false, ah, false, bh, (short)0, acc[ot], false, false);
                acc[ot] = __builtin_amdgcn_wmma_f32_16x16x32_bf16(
                    false, ah, false, bl, (short)0, acc[ot], false, false);
                acc[ot] = __builtin_amdgcn_wmma_f32_16x16x32_bf16(
                    false, al, false, bh, (short)0, acc[ot], false, false);
            }
        }
    }

    // ---- Store D (f32 16x16 layout: VGPR v -> M = laneH*8 + v, N = laneM)
#pragma unroll
    for (int ot = 0; ot < 4; ++ot) {
        const int o = (ot << 4) + laneM;
        const float bv = bias[o];
#pragma unroll
        for (int v = 0; v < 8; ++v) {
            const int j = jbase + laneH * 8 + v;
            out[(((b * HH) + ir) * WW + j) * FF + o] = acc[ot][v] + bv;
        }
    }
}

extern "C" void kernel_launch(void* const* d_in, const int* in_sizes, int n_in,
                              void* d_out, int out_size, void* d_ws, size_t ws_size,
                              hipStream_t stream) {
    const float* x    = (const float*)d_in[0];
    const float* kern = (const float*)d_in[1];
    const float* bias = (const float*)d_in[2];
    float* out = (float*)d_out;

    const size_t lds_bytes = (size_t)LDS_ELEMS * sizeof(__bf16); // ~238.5 KB < 320 KB/WGP
    const dim3 grid(BATCH * HH / 2);   // 256 workgroups, one per 2 output rows
    const dim3 block(256);             // 8 wave32 waves
    fconv3x3_wmma_bf16x3<<<grid, block, lds_bytes, stream>>>(x, kern, bias, out);
}
